// RecurrentHedgeModel_29970281791742
// MI455X (gfx1250) — matrix operand order repaired and checked by reading, not compiled, not run on hardware
//
#include <hip/hip_runtime.h>

// ---------------------------------------------------------------------------
// RecurrentHedgeModel on MI455X (gfx1250, wave32, WMMA).
//
// Compute bound (~470 GFLOP vs ~256 MB HBM traffic). One wave owns 16 batch
// rows for all 30 steps; h lives in WMMA f32 accumulators.
//
// Per step (18 WMMAs):
//   - 10x v_wmma_f32_16x16x32_bf16 : h-dependent parts of Wih/Whh/W1, with
//     (Wih[:,2:34]+Whh) pre-summed for the merged r,z gates.
//   - 8x  v_wmma_f32_16x16x4_f32   : rank-3 chunk [s, delta, 1] x
//     [w_s; w_d; bias] in f32 -- folds all biases into the matmul so every
//     accumulator chain starts from inline-zero C; frags are 2 VGPRs each.
//
// Parity tile remap: tile 0 owns EVEN hidden dims, tile 1 ODD ones (pure
// weight-row reordering at build time; K axis stays natural), so a lane's
// (tile0,tile1) pair lands in adjacent staging columns -> one v_perm pack +
// b32/b64 stores (merged into ds_store_2addr by the compiler).
//
// MLP chain issued first so its relu consumers sit >10 WMMAs away (no
// hazard v_nops); relu via v_med3_f32; delta reduction as a v_pk_add_f32
// tree; relu(a1) pre-scaled by a per-lane W2 pair (no per-step W2 loads);
// S[t] software-pipelined one step ahead. Waves fully independent: private
// LDS slices synced with s_wait_dscnt 0 (per-wave DS ordering), no barriers.
// ---------------------------------------------------------------------------

#define BATCH   1048576
#define STEPS   30
#define HIDDEN  32
#define XDIM    34      // input(1) + delta(1) + hidden(32)
#define WAVES_PER_BLOCK 8
#define ROWS_PER_BLOCK  (WAVES_PER_BLOCK * 16)

typedef __attribute__((ext_vector_type(16))) __bf16    v16bf;
typedef __attribute__((ext_vector_type(8)))  __bf16    v8bf;
typedef __attribute__((ext_vector_type(8)))  float     v8f;
typedef __attribute__((ext_vector_type(2)))  float     v2f;
typedef __attribute__((ext_vector_type(8)))  unsigned  v8u;

// pack bf16(a) into [15:0] and bf16(b) into [31:16], round-half-up
__device__ __forceinline__ unsigned pack_bf16_pair(float a, float b) {
    unsigned ua = __builtin_bit_cast(unsigned, a) + 0x8000u;
    unsigned ub = __builtin_bit_cast(unsigned, b) + 0x8000u;
#if __has_builtin(__builtin_amdgcn_perm)
    return __builtin_amdgcn_perm(ub, ua, 0x07060302u);  // {ub.hi16, ua.hi16}
#else
    return (ua >> 16) | (ub & 0xFFFF0000u);
#endif
}

__device__ __forceinline__ float relu_hw(float x) {
#if __has_builtin(__builtin_amdgcn_fmed3f)
    return __builtin_amdgcn_fmed3f(x, 0.0f, 3.402823466e38f); // v_med3_f32
#else
    return fmaxf(x, 0.0f);
#endif
}

#if __has_builtin(__builtin_amdgcn_tanhf)
__device__ __forceinline__ float tanh_hw(float x) {     // v_tanh_f32 (TRANS)
    return __builtin_amdgcn_tanhf(x);
}
__device__ __forceinline__ float sigmoid_hw(float x) {  // 0.5*tanh(x/2)+0.5
    return __builtin_fmaf(0.5f, __builtin_amdgcn_tanhf(0.5f * x), 0.5f);
}
#else
__device__ __forceinline__ float rcp_hw(float x) {
#if __has_builtin(__builtin_amdgcn_rcpf)
    return __builtin_amdgcn_rcpf(x);
#else
    return 1.0f / x;
#endif
}
__device__ __forceinline__ float tanh_hw(float x) {
    float e = __expf(-2.0f * x);
    return (1.0f - e) * rcp_hw(1.0f + e);
}
__device__ __forceinline__ float sigmoid_hw(float x) {
    return rcp_hw(1.0f + __expf(-x));
}
#endif

__device__ __forceinline__ v8f wmma_bf16(v16bf a, v16bf b, v8f c) {
    return __builtin_amdgcn_wmma_f32_16x16x32_bf16(false, a, false, b,
                                                   (short)0, c, false, false);
}
__device__ __forceinline__ v8f wmma_f32k4(v2f a, v2f b, v8f c) {
    return __builtin_amdgcn_wmma_f32_16x16x4_f32(false, a, false, b,
                                                 (short)0, c, false, false);
}

__global__ void __launch_bounds__(256)
hedge_kernel(const float* __restrict__ S,   const float* __restrict__ W1,
             const float* __restrict__ b1,  const float* __restrict__ W2,
             const float* __restrict__ b2,  const float* __restrict__ Wih,
             const float* __restrict__ bih, const float* __restrict__ Whh,
             const float* __restrict__ bhh, float* __restrict__ out)
{
    // Per-wave private LDS slices (no inter-wave traffic, no barriers).
    __shared__ __align__(16) __bf16 hbuf[WAVES_PER_BLOCK][16][32]; // h transpose
    __shared__ __align__(16) float  abuf[WAVES_PER_BLOCK][16][32]; // scaled relu(a1)

    const int lane = threadIdx.x & 31;
    const int wv   = threadIdx.x >> 5;
    const int lhi  = lane >> 4;          // lane half (0 or 1)
    const int n16  = lane & 15;
    const int rowbase = blockIdx.x * ROWS_PER_BLOCK + wv * 16;

    // ---- bf16 weight fragments, h-dependent K=32 chunk.
    //      B layout: lane holds N=lane&15, elem e -> K = e + 16*lhi.
    //      Parity remap: tile pair (0,1) covers rows (2n, 2n+1).
    //      Built as 8 packed dwords (2 adds + 1 v_perm per PAIR of values).
    v16bf Brz[4];   // (Wih[:,2:34]+Whh) gate rows r,z (merged)
    v16bf Binh[2];  // Wih[:,2:34]       gate rows n
    v16bf Bhn[2];   // Whh               gate rows n
    v16bf B1h[2];   // W1[:,2:34]
#pragma unroll
    for (int j = 0; j < 4; ++j) {
        const int row = (j >> 1) * 32 + 2 * n16 + (j & 1);   // r: 0..31, z: 32..63
        v8u tmp;
#pragma unroll
        for (int d = 0; d < 8; ++d) {
            const int k = 2 * d + 16 * lhi;
            tmp[d] = pack_bf16_pair(
                Wih[row * XDIM + 2 + k]     + Whh[row * HIDDEN + k],
                Wih[row * XDIM + 2 + k + 1] + Whh[row * HIDDEN + k + 1]);
        }
        Brz[j] = __builtin_bit_cast(v16bf, tmp);
    }
#pragma unroll
    for (int j = 0; j < 2; ++j) {
        const int row = 64 + 2 * n16 + j;                    // n-gate rows
        const int r1  = 2 * n16 + j;                         // W1 rows
        v8u ti, th, t1;
#pragma unroll
        for (int d = 0; d < 8; ++d) {
            const int k = 2 * d + 16 * lhi;
            ti[d] = pack_bf16_pair(Wih[row * XDIM + 2 + k],
                                   Wih[row * XDIM + 2 + k + 1]);
            th[d] = pack_bf16_pair(Whh[row * HIDDEN + k],
                                   Whh[row * HIDDEN + k + 1]);
            t1[d] = pack_bf16_pair(W1[r1 * XDIM + 2 + k],
                                   W1[r1 * XDIM + 2 + k + 1]);
        }
        Binh[j] = __builtin_bit_cast(v16bf, ti);
        Bhn[j]  = __builtin_bit_cast(v16bf, th);
        B1h[j]  = __builtin_bit_cast(v16bf, t1);
    }

    // ---- f32 K=4 fragments for the [s, delta, 1] chunk (biases folded in).
    //      A 16x4: lanes 0-15 elem e -> K=e; lanes 16-31 elem e -> K=2+e.
    //      B 4x16: lane holds N=lane&15; same K split. Each frag = 2 VGPRs.
    v2f B4rz[4], B4in[2], B41[2];
#pragma unroll
    for (int j = 0; j < 4; ++j) {
        const int row = (j >> 1) * 32 + 2 * n16 + (j & 1);
        B4rz[j][0] = lhi ? (bih[row] + bhh[row]) : Wih[row * XDIM + 0];
        B4rz[j][1] = lhi ? 0.0f                  : Wih[row * XDIM + 1];
    }
#pragma unroll
    for (int j = 0; j < 2; ++j) {
        const int row = 64 + 2 * n16 + j;
        const int r1  = 2 * n16 + j;
        B4in[j][0] = lhi ? bih[row] : Wih[row * XDIM + 0];
        B4in[j][1] = lhi ? 0.0f     : Wih[row * XDIM + 1];
        B41[j][0]  = lhi ? b1[r1]   : W1[r1 * XDIM + 0];
        B41[j][1]  = lhi ? 0.0f     : W1[r1 * XDIM + 1];
    }
    // h_n bias (inside r*(...), cannot be pre-folded) -> added in gate math.
    float bhnb[2];
#pragma unroll
    for (int j = 0; j < 2; ++j) bhnb[j] = bhh[64 + 2 * n16 + j];
    const float b2v = b2[0];
    // per-lane W2 pair for pre-scaling relu(a1) (parity columns 2n, 2n+1)
    v2f w2p;
    w2p[0] = W2[2 * n16 + 0];
    w2p[1] = W2[2 * n16 + 1];

    // ---- recurrent state: h as two 16x16 f32 C tiles (tile0=even dims,
    //      tile1=odd dims), delta per lane-row.
    v8f h0 = {};
    v8f h1 = {};
    float delta_prev = 0.0f;

    const int k0 = 16 * lhi;       // 16-bit A-layout K offsets (contiguous, matches B)
    const int k1 = 16 * lhi + 8;
    __bf16 (&hb)[16][32] = hbuf[wv];
    float  (&ab)[16][32] = abuf[wv];

    const size_t srow = (size_t)(rowbase + n16) * (STEPS + 1);
    float s_cur = S[srow];                 // software-pipelined S[t]

#pragma unroll 1
    for (int t = 0; t < STEPS; ++t) {
        const float s_next = S[srow + t + 1];   // always in-bounds (31 cols)

        // (1) stage h: pack (even,odd) bf16 pair -> one b32 store per j.
#pragma unroll
        for (int j = 0; j < 8; ++j) {
            const int m = j + 8 * lhi;
            *reinterpret_cast<unsigned*>(&hb[m][2 * n16]) =
                pack_bf16_pair(h0[j], h1[j]);
        }
        asm volatile("s_wait_dscnt 0" ::: "memory"); // per-wave DS in-order

        // (2) A fragments.  bf16 A: lane holds row M=lane&15,
        //     elems 0..7 -> K=k0..k0+7, elems 8..15 -> K=k1..k1+7.
        v8bf lo = *reinterpret_cast<const v8bf*>(&hb[n16][k0]);
        v8bf hi = *reinterpret_cast<const v8bf*>(&hb[n16][k1]);
        v16bf Ah = __builtin_shufflevector(lo, hi, 0, 1, 2, 3, 4, 5, 6, 7,
                                                   8, 9, 10, 11, 12, 13, 14, 15);
        v2f A4;
        A4[0] = lhi ? 1.0f : s_cur;       // K=0: s     | K=2: ones (bias)
        A4[1] = lhi ? 0.0f : delta_prev;  // K=1: delta | K=3: zero

        // (3) 18 WMMAs from inline-zero C.  MLP (a1) chain first so its relu
        //     consumers sit many WMMAs downstream (no hazard nops).
        const v8f zc = {};
        v8f rz[4], inA[2], hnA[2], a1[2];
#pragma unroll
        for (int j = 0; j < 2; ++j) {
            v8f e = wmma_f32k4(A4, B41[j], zc);
            a1[j] = wmma_bf16(Ah, B1h[j], e);
        }
#pragma unroll
        for (int j = 0; j < 4; ++j) {
            v8f c = wmma_f32k4(A4, B4rz[j], zc);
            rz[j] = wmma_bf16(Ah, Brz[j], c);
        }
#pragma unroll
        for (int j = 0; j < 2; ++j) {
            v8f c = wmma_f32k4(A4, B4in[j], zc);
            inA[j] = wmma_bf16(Ah, Binh[j], c);
            hnA[j] = wmma_bf16(Ah, Bhn[j], zc);
        }

        // (4) stage relu(a1)*W2 pair (v_med3 relu + v_pk_mul + b64 store).
#pragma unroll
        for (int j = 0; j < 8; ++j) {
            const int m = j + 8 * lhi;
            v2f p;
            p[0] = relu_hw(a1[0][j]);
            p[1] = relu_hw(a1[1][j]);
            *reinterpret_cast<v2f*>(&ab[m][2 * n16]) = p * w2p;
        }

        // (5) GRU gate math between the ab stores and the dependent
        //     reduction -- hides LDS latency and WMMA->VALU windows.
#pragma unroll
        for (int i = 0; i < 8; ++i) {
            const float r0 = sigmoid_hw(rz[0][i]);
            const float r1 = sigmoid_hw(rz[1][i]);
            const float z0 = sigmoid_hw(rz[2][i]);
            const float z1 = sigmoid_hw(rz[3][i]);
            const float n0 = tanh_hw(__builtin_fmaf(r0, hnA[0][i] + bhnb[0], inA[0][i]));
            const float n1 = tanh_hw(__builtin_fmaf(r1, hnA[1][i] + bhnb[1], inA[1][i]));
            h0[i] = __builtin_fmaf(z0, h0[i] - n0, n0);   // (1-z)*n + z*h
            h1[i] = __builtin_fmaf(z1, h1[i] - n1, n1);
        }

        // (6) delta = sum(ab row) + b2 as a packed v_pk_add_f32 tree.
        asm volatile("s_wait_dscnt 0" ::: "memory");
        const v2f* pp = reinterpret_cast<const v2f*>(&ab[n16][0]);
        v2f q0 = (pp[0]  + pp[1])  + (pp[2]  + pp[3]);
        v2f q1 = (pp[4]  + pp[5])  + (pp[6]  + pp[7]);
        v2f q2 = (pp[8]  + pp[9])  + (pp[10] + pp[11]);
        v2f q3 = (pp[12] + pp[13]) + (pp[14] + pp[15]);
        v2f qa = (q0 + q1) + (q2 + q3);
        const float sum = qa[0] + qa[1] + b2v;
        delta_prev = sum;                 // lane L holds delta for row L&15
        if (lane < 16)
            out[(size_t)(rowbase + lane) * STEPS + t] = sum;
        s_cur = s_next;
    }
}

extern "C" void kernel_launch(void* const* d_in, const int* in_sizes, int n_in,
                              void* d_out, int out_size, void* d_ws, size_t ws_size,
                              hipStream_t stream) {
    (void)in_sizes; (void)n_in; (void)out_size; (void)d_ws; (void)ws_size;
    const float* S   = (const float*)d_in[0];
    const float* W1  = (const float*)d_in[1];
    const float* b1  = (const float*)d_in[2];
    const float* W2  = (const float*)d_in[3];
    const float* b2  = (const float*)d_in[4];
    const float* Wih = (const float*)d_in[5];
    const float* bih = (const float*)d_in[6];
    const float* Whh = (const float*)d_in[7];
    const float* bhh = (const float*)d_in[8];
    float* out = (float*)d_out;

    dim3 grid(BATCH / ROWS_PER_BLOCK);   // 8192 blocks x 8 waves x 16 rows
    dim3 block(256);
    hipLaunchKernelGGL(hedge_kernel, grid, block, 0, stream,
                       S, W1, b1, W2, b2, Wih, bih, Whh, bhh, out);
}
